// MultiHeadAttention_37538014167850
// MI455X (gfx1250) — matrix operand order repaired and checked by reading, not compiled, not run on hardware
//
#include <hip/hip_runtime.h>
#include <hip/hip_bf16.h>

// ---- problem constants (from reference) ----
constexpr int Bc  = 2;
constexpr int Sc  = 2048;
constexpr int Dc  = 1024;
constexpr int Hc  = 16;
constexpr int DKc = 64;   // head dim
constexpr size_t BSD = (size_t)Bc * Sc * Dc;   // 4,194,304 elements
constexpr size_t DD  = (size_t)Dc * Dc;        // 1,048,576 elements

typedef __attribute__((ext_vector_type(16))) __bf16 v16bf;
typedef __attribute__((ext_vector_type(8)))  __bf16 v8bf;
typedef __attribute__((ext_vector_type(8)))  float  v8f;
typedef int gv4i __attribute__((vector_size(4 * sizeof(int))));

#ifndef __has_builtin
#define __has_builtin(x) 0
#endif
#if __has_builtin(__builtin_amdgcn_global_load_async_to_lds_b128)
#define ASYNC_LDS 1
#endif

__device__ __forceinline__ unsigned short f32_bf16(float f) {
    union { float f; unsigned u; } x{f};
    unsigned u = x.u;
    unsigned r = (u + 0x7FFFu + ((u >> 16) & 1u)) >> 16;   // RNE
    return (unsigned short)r;
}

__device__ __forceinline__ v16bf cat8(v8bf lo, v8bf hi) {
    return __builtin_shufflevector(lo, hi, 0,1,2,3,4,5,6,7,8,9,10,11,12,13,14,15);
}

__device__ __forceinline__ v8f wmma_bf16(v16bf a, v16bf b, v8f c) {
    return __builtin_amdgcn_wmma_f32_16x16x32_bf16(
        false, a, false, b, (short)0, c, false, false);
}

// 16-byte global -> LDS copy; async path on CDNA5, sync fallback otherwise
__device__ __forceinline__ void async_copy16(const unsigned short* g,
                                             unsigned short* l) {
#if ASYNC_LDS
    __builtin_amdgcn_global_load_async_to_lds_b128(
        (__attribute__((address_space(1))) gv4i*)(unsigned long long)g,
        (__attribute__((address_space(3))) gv4i*)(unsigned)(unsigned long long)l,
        0, 0);
#else
    *(uint4*)l = *(const uint4*)g;
#endif
}

__device__ __forceinline__ void async_wait() {
#if ASYNC_LDS
#if __has_builtin(__builtin_amdgcn_s_wait_asynccnt)
    __builtin_amdgcn_s_wait_asynccnt(0);
#else
    asm volatile("s_wait_asynccnt 0x0" ::: "memory");
#endif
#endif
}

// ---------------------------------------------------------------------------
// fp32 -> bf16 elementwise conversion
// ---------------------------------------------------------------------------
__global__ void cvt_f32_bf16_kernel(const float* __restrict__ in,
                                    unsigned short* __restrict__ out, int n) {
    int i = blockIdx.x * blockDim.x + threadIdx.x;
    int stride = gridDim.x * blockDim.x;
    for (; i < n; i += stride) out[i] = f32_bf16(in[i]);
}

// ---------------------------------------------------------------------------
// GEMM: out = A(MxK bf16) @ W(NxK bf16)^T + bias
// Block = 256 threads (8 waves) -> 128x128 tile, LDS-staged, K-step 32.
// Wave grid 4x2: each wave 32 rows x 64 cols (8 WMMA accumulators).
// ---------------------------------------------------------------------------
__launch_bounds__(256, 2)
__global__ void gemm_bias_kernel(const unsigned short* __restrict__ A,
                                 const unsigned short* __restrict__ W,
                                 const float* __restrict__ bias,
                                 unsigned short* __restrict__ outb,
                                 float* __restrict__ outf,
                                 int M, int N, int K) {
    __shared__ unsigned short As[128 * 32];   // [row][k] 8KB
    __shared__ unsigned short Bs[128 * 32];   // [col][k] 8KB
    const int tid  = threadIdx.x;
    const int lane = tid & 31, warp = tid >> 5;
    const int nblk = N >> 7;
    const int m0 = (blockIdx.x / nblk) << 7;
    const int n0 = (blockIdx.x % nblk) << 7;
    const int wrow = warp >> 1, wcol = warp & 1;
    const int half = lane >> 4, ml = lane & 15;

    v8f acc[2][4] = {};

    // copy mapping: 512 16B-chunks per tile, 2 per thread (4 chunks per row)
    const int r0 = tid >> 2,          o0c = (tid & 3) * 8;
    const int r1 = (tid + 256) >> 2,  o1c = ((tid + 256) & 3) * 8;

    for (int kk = 0; kk < K; kk += 32) {
        __syncthreads();   // previous iteration's LDS reads complete
        async_copy16(A + (size_t)(m0 + r0) * K + kk + o0c, As + r0 * 32 + o0c);
        async_copy16(A + (size_t)(m0 + r1) * K + kk + o1c, As + r1 * 32 + o1c);
        async_copy16(W + (size_t)(n0 + r0) * K + kk + o0c, Bs + r0 * 32 + o0c);
        async_copy16(W + (size_t)(n0 + r1) * K + kk + o1c, Bs + r1 * 32 + o1c);
        async_wait();
        __syncthreads();   // tiles visible to all waves
#pragma unroll
        for (int ms = 0; ms < 2; ++ms) {
            const unsigned short* ap = As + (wrow * 32 + ms * 16 + ml) * 32 + half * 16;
            v16bf af = *(const v16bf*)ap;
#pragma unroll
            for (int nt = 0; nt < 4; ++nt) {
                const unsigned short* bp =
                    Bs + (wcol * 64 + nt * 16 + ml) * 32 + half * 16;
                acc[ms][nt] = wmma_bf16(af, *(const v16bf*)bp, acc[ms][nt]);
            }
        }
    }

#pragma unroll
    for (int ms = 0; ms < 2; ++ms)
#pragma unroll
        for (int nt = 0; nt < 4; ++nt) {
            const int col = n0 + wcol * 64 + nt * 16 + ml;
            const float bv = bias[col];
#pragma unroll
            for (int r = 0; r < 8; ++r) {
                const int row = m0 + wrow * 32 + ms * 16 + r + half * 8;
                const float v = acc[ms][nt][r] + bv;
                if (outf) outf[(size_t)row * N + col] = v;
                else      outb[(size_t)row * N + col] = f32_bf16(v);
            }
        }
}

// ---------------------------------------------------------------------------
// V (B,S,H*DK) bf16 -> Vt (B,H,DK,S) bf16 so PV B-fragments load contiguously
// ---------------------------------------------------------------------------
__global__ void transpose_v_kernel(const unsigned short* __restrict__ Vp,
                                   unsigned short* __restrict__ Vt) {
    size_t i = (size_t)blockIdx.x * blockDim.x + threadIdx.x;
    if (i >= BSD) return;
    int s = (int)(i % Sc); size_t t = i / Sc;
    int d = (int)(t % DKc); t /= DKc;
    int h = (int)(t % Hc);  int b = (int)(t / Hc);
    Vt[i] = Vp[((size_t)b * Sc + s) * Dc + h * DKc + d];
}

// ---------------------------------------------------------------------------
// Causal flash attention. Block = 8 waves = 128 consecutive queries of one
// (b,h); K/V 32-key blocks staged in LDS once and shared by all 8 waves.
// ---------------------------------------------------------------------------
__launch_bounds__(256, 2)
__global__ void attn_kernel(const unsigned short* __restrict__ Q,
                            const unsigned short* __restrict__ Kb,
                            const unsigned short* __restrict__ Vt,
                            unsigned short* __restrict__ O) {
    __shared__ unsigned short Ks[32 * 64];     // [key][d]   4KB
    __shared__ unsigned short Vs[64 * 32];     // [d][key]   4KB
    __shared__ unsigned short Ps[8][16 * 32];  // per-wave P 8KB

    const int tid  = threadIdx.x;
    const int lane = tid & 31, warp = tid >> 5;
    const int qgroups = Sc / 128;              // 16
    const int qg = blockIdx.x % qgroups;
    int bh = blockIdx.x / qgroups;
    const int h = bh % Hc, b = bh / Hc;
    const int qg0 = qg * 128;
    const int q0  = qg0 + warp * 16;
    const int half = lane >> 4, ml = lane & 15;

    // Q A-fragments (d-chunks [0,32) and [32,64)) straight from global
    const unsigned short* qrow =
        Q + ((size_t)(b * Sc + q0 + ml)) * Dc + h * DKc + half * 16;
    v16bf aq0 = *(const v16bf*)(qrow);
    v16bf aq1 = *(const v16bf*)(qrow + 32);

    float mrow[8], lrow[8];
#pragma unroll
    for (int r = 0; r < 8; ++r) { mrow[r] = -3.0e38f; lrow[r] = 0.0f; }
    v8f o0 = {}, o1 = {}, o2 = {}, o3 = {};
    const float scale = 0.125f;                // 1/sqrt(64)
    unsigned short* myp = Ps[warp];

    // staging maps: K tile = 32 rows x 128B (8 chunks/row); V = 64 rows x 64B
    const int krow = tid >> 3, koff = (tid & 7) * 8;
    const int vrow = tid >> 2, voff = (tid & 3) * 8;
    const unsigned short* kbase = Kb + (size_t)b * Sc * Dc + h * DKc;
    const unsigned short* vbase = Vt + ((size_t)(b * Hc + h)) * DKc * Sc;

    const int jend = qg0 + 112;                // covers keys up to qg0+127
    for (int j = 0; j <= jend; j += 32) {
        __syncthreads();
        async_copy16(kbase + (size_t)(j + krow) * Dc + koff, Ks + krow * 64 + koff);
        async_copy16(vbase + (size_t)vrow * Sc + j + voff,   Vs + vrow * 32 + voff);
        async_wait();
        __syncthreads();

        // ---- scores S = Q K^T from LDS ----
        const unsigned short* kp = Ks + ml * 64 + half * 16;
        v8f s0 = {}, s1 = {};
        s0 = wmma_bf16(aq0, *(const v16bf*)(kp),            s0);
        s1 = wmma_bf16(aq0, *(const v16bf*)(kp + 16 * 64),  s1);
        s0 = wmma_bf16(aq1, *(const v16bf*)(kp + 32),       s0);
        s1 = wmma_bf16(aq1, *(const v16bf*)(kp + 16 * 64 + 32), s1);

        // ---- online softmax (rows = VGPR idx + 8*half) ----
#pragma unroll
        for (int r = 0; r < 8; ++r) {
            const int qi = q0 + r + half * 8;
            float v0 = s0[r] * scale; if (j + ml > qi)      v0 = -1.0e9f;
            float v1 = s1[r] * scale; if (j + 16 + ml > qi) v1 = -1.0e9f;
            float rm = fmaxf(v0, v1);
            rm = fmaxf(rm, __shfl_xor(rm, 1, 32));
            rm = fmaxf(rm, __shfl_xor(rm, 2, 32));
            rm = fmaxf(rm, __shfl_xor(rm, 4, 32));
            rm = fmaxf(rm, __shfl_xor(rm, 8, 32));
            const float mn    = fmaxf(mrow[r], rm);
            const float alpha = __expf(mrow[r] - mn);
            const float p0 = __expf(v0 - mn), p1 = __expf(v1 - mn);
            float rs = p0 + p1;
            rs += __shfl_xor(rs, 1, 32);
            rs += __shfl_xor(rs, 2, 32);
            rs += __shfl_xor(rs, 4, 32);
            rs += __shfl_xor(rs, 8, 32);
            lrow[r] = lrow[r] * alpha + rs;
            mrow[r] = mn;
            o0[r] *= alpha; o1[r] *= alpha; o2[r] *= alpha; o3[r] *= alpha;
            const int prow = r + half * 8;
            myp[prow * 32 + ml]      = f32_bf16(p0);
            myp[prow * 32 + 16 + ml] = f32_bf16(p1);
        }

        // ---- reload P as A-fragment (LDS in-order per wave) ----
        const unsigned short* pr = myp + ml * 32 + half * 16;
        v16bf pf = *(const v16bf*)(pr);

        // ---- O += P V from LDS V tile ----
        const unsigned short* vp = Vs + ml * 32 + half * 16;
        o0 = wmma_bf16(pf, *(const v16bf*)(vp),           o0);
        o1 = wmma_bf16(pf, *(const v16bf*)(vp + 16 * 32), o1);
        o2 = wmma_bf16(pf, *(const v16bf*)(vp + 32 * 32), o2);
        o3 = wmma_bf16(pf, *(const v16bf*)(vp + 48 * 32), o3);
    }

    // ---- epilogue: normalize, store bf16 (B,S,D) for final projection ----
#pragma unroll
    for (int r = 0; r < 8; ++r) {
        const int qi = q0 + r + half * 8;
        const float inv = 1.0f / lrow[r];
        const size_t base = ((size_t)(b * Sc + qi)) * Dc + h * DKc + ml;
        O[base]      = f32_bf16(o0[r] * inv);
        O[base + 16] = f32_bf16(o1[r] * inv);
        O[base + 32] = f32_bf16(o2[r] * inv);
        O[base + 48] = f32_bf16(o3[r] * inv);
    }
}

// ---------------------------------------------------------------------------
// launch
// ---------------------------------------------------------------------------
extern "C" void kernel_launch(void* const* d_in, const int* in_sizes, int n_in,
                              void* d_out, int out_size, void* d_ws, size_t ws_size,
                              hipStream_t stream) {
    const float* q  = (const float*)d_in[0];
    const float* k  = (const float*)d_in[1];
    const float* v  = (const float*)d_in[2];
    // d_in[3] = mask: causal, computed analytically -> not read
    const float* Wq = (const float*)d_in[4];  const float* bq = (const float*)d_in[5];
    const float* Wk = (const float*)d_in[6];  const float* bk = (const float*)d_in[7];
    const float* Wv = (const float*)d_in[8];  const float* bv = (const float*)d_in[9];
    const float* Wo = (const float*)d_in[10]; const float* bo = (const float*)d_in[11];

    unsigned short* ws = (unsigned short*)d_ws;
    size_t off = 0;
    unsigned short* qb  = ws + off; off += BSD;
    unsigned short* kb  = ws + off; off += BSD;
    unsigned short* vb  = ws + off; off += BSD;
    unsigned short* wqb = ws + off; off += DD;
    unsigned short* wkb = ws + off; off += DD;
    unsigned short* wvb = ws + off; off += DD;
    unsigned short* wob = ws + off; off += DD;
    unsigned short* Qp  = ws + off; off += BSD;
    unsigned short* Kp  = ws + off; off += BSD;
    unsigned short* Vp  = ws + off; off += BSD;
    unsigned short* Vt  = ws + off; off += BSD;
    unsigned short* Ao  = ws + off; off += BSD;

    // 1) fp32 -> bf16 staging
    cvt_f32_bf16_kernel<<<(int)(BSD / 256), 256, 0, stream>>>(q,  qb,  (int)BSD);
    cvt_f32_bf16_kernel<<<(int)(BSD / 256), 256, 0, stream>>>(k,  kb,  (int)BSD);
    cvt_f32_bf16_kernel<<<(int)(BSD / 256), 256, 0, stream>>>(v,  vb,  (int)BSD);
    cvt_f32_bf16_kernel<<<(int)(DD  / 256), 256, 0, stream>>>(Wq, wqb, (int)DD);
    cvt_f32_bf16_kernel<<<(int)(DD  / 256), 256, 0, stream>>>(Wk, wkb, (int)DD);
    cvt_f32_bf16_kernel<<<(int)(DD  / 256), 256, 0, stream>>>(Wv, wvb, (int)DD);
    cvt_f32_bf16_kernel<<<(int)(DD  / 256), 256, 0, stream>>>(Wo, wob, (int)DD);

    // 2) Q/K/V projections: M=4096, N=K=1024 -> (4096/128)*(1024/128) = 256 blocks
    const int M = Bc * Sc, N = Dc, K = Dc;
    const int gemm_blocks = (M / 128) * (N / 128);
    gemm_bias_kernel<<<gemm_blocks, 256, 0, stream>>>(qb, wqb, bq, Qp, nullptr, M, N, K);
    gemm_bias_kernel<<<gemm_blocks, 256, 0, stream>>>(kb, wkb, bk, Kp, nullptr, M, N, K);
    gemm_bias_kernel<<<gemm_blocks, 256, 0, stream>>>(vb, wvb, bv, Vp, nullptr, M, N, K);

    // 3) V -> (B,H,DK,S)
    transpose_v_kernel<<<(int)(BSD / 256), 256, 0, stream>>>(Vp, Vt);

    // 4) causal attention: B*H*(S/128) = 512 blocks of 8 waves
    attn_kernel<<<Bc * Hc * (Sc / 128), 256, 0, stream>>>(Qp, Kp, Vt, Ao);

    // 5) output projection -> fp32 d_out
    gemm_bias_kernel<<<gemm_blocks, 256, 0, stream>>>(Ao, wob, bo, nullptr, (float*)d_out, M, N, K);
}
